// CreditCardMultiHeadAttention_86818468921450
// MI455X (gfx1250) — matrix operand order repaired and checked by reading, not compiled, not run on hardware
//
#include <hip/hip_runtime.h>
#include <hip/hip_bf16.h>
#include <cstdint>

// ---------------------------------------------------------------------------
// Multi-head attention forward, CDNA5 (gfx1250), wave32 + WMMA bf16 +
// async global->LDS copies (ASYNCcnt) with double-buffered LDS tiles.
//   D_MODEL=1024, H=16, Dh=64, B=2, S=2048.
// Pipeline:
//   0) cvt_bf16 / transpose_bf16 : x -> bf16, W -> bf16 W^T (one-time, ~25MB)
//   1) gemm_wmma<0,bf16> x2      : Q,K projections -> bf16 [tok, d]
//      gemm_wmma<1,bf16>         : V projection    -> bf16 V^T [b,h,d,s]
//   2) flash_attn_wmma           : fused QK^T -> online softmax -> P.V
//   3) gemm_wmma<0,float>        : ctx @ Wo^T + bo -> fp32 d_out
// Workspace: xb 8MB + 4x W^T 2MB + q/k/vT/ctx 4x8MB = 48 MB.
// ---------------------------------------------------------------------------

#define D_MODEL   1024
#define NUM_HEADS 16
#define HEAD_DIM  64
#define BATCH     2
#define SEQ       2048

typedef __bf16 v16bf __attribute__((ext_vector_type(16)));
typedef float  v8f   __attribute__((ext_vector_type(8)));

// ---- CDNA5 async copy primitives (inline asm, toolchain-portable) ---------
__device__ __forceinline__ unsigned lds_off32(const void* p) {
  // Flat LDS pointers keep the LDS byte offset in addr[31:0] (ISA 10.2).
  return (unsigned)(uintptr_t)p;
}
__device__ __forceinline__ void async_cp16(unsigned lds_byte, const void* gbase,
                                           unsigned goff_bytes) {
  // GVS mode: LDS[lds_byte] <= MEM[sgpr_base + vgpr_off], 16 bytes, ASYNCcnt++
  asm volatile("global_load_async_to_lds_b128 %0, %1, %2"
               :: "v"(lds_byte), "v"(goff_bytes), "s"(gbase)
               : "memory");
}
#define ASYNC_WAIT(imm) asm volatile("s_wait_asynccnt " #imm ::: "memory")

// ---- WMMA fragment helpers (ISA 7.12.2 layouts) ---------------------------
// A (16x32 bf16): lane l -> row l%16, kb=(l/16)*8; e<8: K=kb+e, e>=8: K=16+kb+e
__device__ __forceinline__ v16bf load_a_frag(const __bf16* row, int kb) {
  v16bf a;
#pragma unroll
  for (int e = 0; e < 8; ++e) { a[e] = row[kb + e]; a[8 + e] = row[16 + kb + e]; }
  return a;
}
// B (32x16 bf16) from K-major (transposed) tile: lane l -> col l%16, kb=kb2/2; e<8: K=kb+e, e>=8: K=16+kb+e
__device__ __forceinline__ v16bf load_b_frag(const __bf16* coln, int kb2) {
  v16bf b;
#pragma unroll
  for (int e = 0; e < 8; ++e) { b[e] = coln[kb2 / 2 + e]; b[8 + e] = coln[16 + kb2 / 2 + e]; }
  return b;
}
__device__ __forceinline__ v8f wmma_bf16(v16bf a, v16bf b, v8f c) {
  return __builtin_amdgcn_wmma_f32_16x16x32_bf16(false, a, false, b,
                                                 (short)0, c, false, false);
}

// ---------------------------------------------------------------------------
// One-time format kernels.
// ---------------------------------------------------------------------------
__global__ void __launch_bounds__(256)
cvt_bf16(const float* __restrict__ in, __bf16* __restrict__ out, int n) {
  int i = blockIdx.x * 256 + threadIdx.x;
  if (i < n) out[i] = (__bf16)in[i];
}

// Wt[c][r] = (bf16)W[r][c], both 1024x1024; 32x32 LDS tile, 256 threads.
__global__ void __launch_bounds__(256)
transpose_bf16(const float* __restrict__ W, __bf16* __restrict__ Wt,
               int R, int C) {
  __shared__ float t[32][33];
  int r0 = blockIdx.y * 32, c0 = blockIdx.x * 32;
  int tx = threadIdx.x & 31, ty = threadIdx.x >> 5;
  for (int i = ty; i < 32; i += 8)
    t[i][tx] = W[(size_t)(r0 + i) * C + c0 + tx];
  __syncthreads();
  for (int i = ty; i < 32; i += 8)
    Wt[(size_t)(c0 + i) * R + r0 + tx] = (__bf16)t[tx][i];
}

// ---------------------------------------------------------------------------
// C = A(MxK,bf16) * Bt^T + bias   (Bt is N x K, pre-transposed weights).
// 128 threads / 4 waves, 64x64 tile, BK=32, double-buffered async LDS.
// OUT_LAYOUT 0: row-major [M,N].  OUT_LAYOUT 1: per-head V^T [b,h,d,s].
// ---------------------------------------------------------------------------
template <int OUT_LAYOUT, typename OT>
__global__ void __launch_bounds__(128)
gemm_wmma(const __bf16* __restrict__ A, const __bf16* __restrict__ Bt,
          const float* __restrict__ bias, OT* __restrict__ C,
          int M, int N, int K) {
  __shared__ __align__(16) __bf16 As[2][64][32];
  __shared__ __align__(16) __bf16 Bs[2][64][32];

  const int tid  = threadIdx.x;
  const int wave = tid >> 5, lane = tid & 31;
  const int lr   = lane & 15, half = lane >> 4;
  const int kb   = half * 8, kb2 = half * 16;
  const int m0 = blockIdx.y * 64, n0 = blockIdx.x * 64;
  const int mw = (wave >> 1) * 32, nw = (wave & 1) * 32;

  const unsigned asB = lds_off32(&As[0][0][0]);
  const unsigned bsB = lds_off32(&Bs[0][0][0]);

  auto issue_tile = [&](int kt, int buf) {  // 4 async b128 per thread
    const int k0 = kt * 32;
#pragma unroll
    for (int it = 0; it < 2; ++it) {
      unsigned off = (unsigned)(tid + it * 128) * 16;   // tile byte offset
      unsigned r = off >> 6, cb = off & 63;             // 64B per 32-elem row
      async_cp16(asB + buf * 4096u + off, A,
                 (unsigned)((m0 + r) * K + k0) * 2u + cb);
      async_cp16(bsB + buf * 4096u + off, Bt,
                 (unsigned)((n0 + r) * K + k0) * 2u + cb);
    }
  };

  v8f acc[2][2];
#pragma unroll
  for (int mi = 0; mi < 2; ++mi)
#pragma unroll
    for (int ni = 0; ni < 2; ++ni) acc[mi][ni] = (v8f){};

  const int nK = K / 32;
  issue_tile(0, 0);
  for (int kt = 0; kt < nK; ++kt) {
    const int cur = kt & 1;
    if (kt + 1 < nK) { issue_tile(kt + 1, cur ^ 1); ASYNC_WAIT(0x4); }
    else             { ASYNC_WAIT(0x0); }
    __syncthreads();

    v16bf af[2], bfg[2];
#pragma unroll
    for (int mi = 0; mi < 2; ++mi) af[mi]  = load_a_frag(&As[cur][mw + 16 * mi + lr][0], kb);
#pragma unroll
    for (int ni = 0; ni < 2; ++ni) bfg[ni] = load_b_frag(&Bs[cur][nw + 16 * ni + lr][0], kb2);
#pragma unroll
    for (int mi = 0; mi < 2; ++mi)
#pragma unroll
      for (int ni = 0; ni < 2; ++ni)
        acc[mi][ni] = wmma_bf16(af[mi], bfg[ni], acc[mi][ni]);
    __syncthreads();
  }

  // C/D layout: M = j + 8*(lane/16), N = lane%16
#pragma unroll
  for (int mi = 0; mi < 2; ++mi)
#pragma unroll
    for (int ni = 0; ni < 2; ++ni)
#pragma unroll
      for (int jj = 0; jj < 8; ++jj) {
        int gr = m0 + mw + 16 * mi + jj + 8 * half;
        int gc = n0 + nw + 16 * ni + lr;
        float v = acc[mi][ni][jj] + bias[gc];
        if constexpr (OUT_LAYOUT == 0) {
          C[(size_t)gr * N + gc] = (OT)v;
        } else {               // V^T: [b, h, d, s]
          int bb = gr >> 11, ss = gr & (SEQ - 1);
          int hh = gc >> 6,  dd = gc & (HEAD_DIM - 1);
          C[((size_t)(bb * NUM_HEADS + hh) * HEAD_DIM + dd) * SEQ + ss] = (OT)v;
        }
      }
}

// ---------------------------------------------------------------------------
// Fused flash attention. Block = (q-block 64, head, batch), 4 waves.
// K tile: natural [kv][d]; V tile: from pre-transposed V^T -> Vs[d][kv].
// Both streamed with double-buffered async copies.
// ---------------------------------------------------------------------------
__global__ void __launch_bounds__(128)
flash_attn_wmma(const __bf16* __restrict__ Q, const __bf16* __restrict__ Kg,
                const __bf16* __restrict__ VT,
                const unsigned char* __restrict__ mask,
                __bf16* __restrict__ Ctx) {
  __shared__ __align__(16) __bf16 Ks[2][64][64];
  __shared__ __align__(16) __bf16 Vs[2][64][64];
  __shared__ __align__(16) __bf16 Ps[4][16][64];

  const int tid  = threadIdx.x;
  const int wave = tid >> 5, lane = tid & 31;
  const int lr   = lane & 15, half = lane >> 4;
  const int kb   = half * 8, kb2 = half * 16;
  const int qbase = blockIdx.x * 64;
  const int h = blockIdx.y, b = blockIdx.z;
  const float sc = 0.125f;                       // 1/sqrt(HEAD_DIM)

  const unsigned ksB = lds_off32(&Ks[0][0][0]);
  const unsigned vsB = lds_off32(&Vs[0][0][0]);
  const char* kBase = (const char*)Kg +
      ((size_t)b * SEQ * D_MODEL + h * HEAD_DIM) * 2;
  const char* vBase = (const char*)VT +
      ((size_t)(b * NUM_HEADS + h) * HEAD_DIM) * SEQ * 2;

  auto issue_kv = [&](int j, int buf) {          // 8 async b128 per thread
#pragma unroll
    for (int it = 0; it < 4; ++it) {
      unsigned off = (unsigned)(tid + it * 128) * 16;  // 0..8176
      unsigned r = off >> 7, cb = off & 127;           // 128B per 64-elem row
      async_cp16(ksB + buf * 8192u + off, kBase,
                 (unsigned)(j * 64 + r) * (D_MODEL * 2u) + cb);
      async_cp16(vsB + buf * 8192u + off, vBase,
                 r * (SEQ * 2u) + (unsigned)j * 128u + cb);
    }
  };

  // Preload this wave's Q A-fragments (row = lane%16, 2 k-steps of d).
  const int qrow = qbase + wave * 16 + lr;
  const __bf16* qrp = Q + (size_t)(b * SEQ + qrow) * D_MODEL + h * HEAD_DIM;
  v16bf qa[2];
#pragma unroll
  for (int ks = 0; ks < 2; ++ks) qa[ks] = load_a_frag(qrp + ks * 32, kb);

  float mprev[8], lprev[8];
  v8f o[4];
#pragma unroll
  for (int jj = 0; jj < 8; ++jj) { mprev[jj] = -1e30f; lprev[jj] = 0.0f; }
#pragma unroll
  for (int ni = 0; ni < 4; ++ni) o[ni] = (v8f){};

  issue_kv(0, 0);
  for (int j = 0; j < SEQ / 64; ++j) {
    const int cur = j & 1;
    if (j + 1 < SEQ / 64) { issue_kv(j + 1, cur ^ 1); ASYNC_WAIT(0x8); }
    else                  { ASYNC_WAIT(0x0); }
    __syncthreads();

    // S = Q K^T : B^T tile is Ks in natural [kv][d] layout.
    v8f s[4];
#pragma unroll
    for (int ni = 0; ni < 4; ++ni) s[ni] = (v8f){};
#pragma unroll
    for (int ks = 0; ks < 2; ++ks)
#pragma unroll
      for (int ni = 0; ni < 4; ++ni) {
        v16bf bk = load_b_frag(&Ks[cur][ni * 16 + lr][ks * 32], kb2);
        s[ni] = wmma_bf16(qa[ks], bk, s[ni]);
      }

    float mb[4];
#pragma unroll
    for (int ni = 0; ni < 4; ++ni) {
      int col = j * 64 + ni * 16 + lr;
      mb[ni] = mask[b * SEQ + col] ? 0.0f : -1e30f;
    }

    // Online softmax; row = jj + 8*half lives in 16 lanes of one half.
#pragma unroll
    for (int jj = 0; jj < 8; ++jj) {
      float rm = -1e30f;
#pragma unroll
      for (int ni = 0; ni < 4; ++ni) rm = fmaxf(rm, s[ni][jj] * sc + mb[ni]);
      for (int off = 1; off < 16; off <<= 1) rm = fmaxf(rm, __shfl_xor(rm, off, 32));
      float mnew = fmaxf(mprev[jj], rm);
      float corr = __expf(mprev[jj] - mnew);
      float sum = 0.0f;
#pragma unroll
      for (int ni = 0; ni < 4; ++ni) {
        float p = __expf(s[ni][jj] * sc + mb[ni] - mnew);
        s[ni][jj] = p;
        sum += p;
      }
      for (int off = 1; off < 16; off <<= 1) sum += __shfl_xor(sum, off, 32);
      lprev[jj] = lprev[jj] * corr + sum;
      mprev[jj] = mnew;
#pragma unroll
      for (int ni = 0; ni < 4; ++ni) o[ni][jj] *= corr;
    }

    // Stage P (C-layout regs -> row-major LDS) for A-layout reload.
#pragma unroll
    for (int jj = 0; jj < 8; ++jj) {
      int r = jj + 8 * half;
#pragma unroll
      for (int ni = 0; ni < 4; ++ni)
        Ps[wave][r][ni * 16 + lr] = (__bf16)s[ni][jj];
    }

    // O += P @ V : B^T tile is Vs (= V transposed).
#pragma unroll
    for (int ks = 0; ks < 2; ++ks) {
      v16bf ap = load_a_frag(&Ps[wave][lr][ks * 32], kb);
#pragma unroll
      for (int ni = 0; ni < 4; ++ni) {
        v16bf bv = load_b_frag(&Vs[cur][ni * 16 + lr][ks * 32], kb2);
        o[ni] = wmma_bf16(ap, bv, o[ni]);
      }
    }
    __syncthreads();
  }

  // Normalize, write context in natural [B,S,H*Dh] layout (bf16).
#pragma unroll
  for (int jj = 0; jj < 8; ++jj) {
    int row = qbase + wave * 16 + jj + 8 * half;
    float inv = 1.0f / lprev[jj];
#pragma unroll
    for (int ni = 0; ni < 4; ++ni) {
      int d = ni * 16 + lr;
      Ctx[(size_t)(b * SEQ + row) * D_MODEL + h * HEAD_DIM + d] =
          (__bf16)(o[ni][jj] * inv);
    }
  }
}

// ---------------------------------------------------------------------------
extern "C" void kernel_launch(void* const* d_in, const int* in_sizes, int n_in,
                              void* d_out, int out_size, void* d_ws, size_t ws_size,
                              hipStream_t stream) {
  const float*         x    = (const float*)d_in[0];
  const unsigned char* mask = (const unsigned char*)d_in[1];
  const float* Wq = (const float*)d_in[2];
  const float* bq = (const float*)d_in[3];
  const float* Wk = (const float*)d_in[4];
  const float* bk = (const float*)d_in[5];
  const float* Wv = (const float*)d_in[6];
  const float* bv = (const float*)d_in[7];
  const float* Wo = (const float*)d_in[8];
  const float* bo = (const float*)d_in[9];
  float* out = (float*)d_out;

  const size_t NTOK = (size_t)BATCH * SEQ;              // 4096
  const size_t DSQ  = (size_t)D_MODEL * D_MODEL;        // 1M
  __bf16* xb   = (__bf16*)d_ws;                         // 8 MB
  __bf16* wqT  = xb   + NTOK * D_MODEL;                 // 2 MB each
  __bf16* wkT  = wqT  + DSQ;
  __bf16* wvT  = wkT  + DSQ;
  __bf16* woT  = wvT  + DSQ;
  __bf16* qbuf = woT  + DSQ;                            // 8 MB each
  __bf16* kbuf = qbuf + NTOK * D_MODEL;
  __bf16* vTbf = kbuf + NTOK * D_MODEL;
  __bf16* cbuf = vTbf + NTOK * D_MODEL;                 // total 48 MB

  // 0) one-time format conversion
  const int nconv = (int)(NTOK * D_MODEL);
  cvt_bf16<<<nconv / 256, 256, 0, stream>>>(x, xb, nconv);
  dim3 tg(D_MODEL / 32, D_MODEL / 32), tb(256);
  transpose_bf16<<<tg, tb, 0, stream>>>(Wq, wqT, D_MODEL, D_MODEL);
  transpose_bf16<<<tg, tb, 0, stream>>>(Wk, wkT, D_MODEL, D_MODEL);
  transpose_bf16<<<tg, tb, 0, stream>>>(Wv, wvT, D_MODEL, D_MODEL);
  transpose_bf16<<<tg, tb, 0, stream>>>(Wo, woT, D_MODEL, D_MODEL);

  // 1) projections
  dim3 blk(128);
  dim3 gg(D_MODEL / 64, (unsigned)(NTOK / 64));         // (16, 64)
  gemm_wmma<0, __bf16><<<gg, blk, 0, stream>>>(xb, wqT, bq, qbuf,
                                               (int)NTOK, D_MODEL, D_MODEL);
  gemm_wmma<0, __bf16><<<gg, blk, 0, stream>>>(xb, wkT, bk, kbuf,
                                               (int)NTOK, D_MODEL, D_MODEL);
  gemm_wmma<1, __bf16><<<gg, blk, 0, stream>>>(xb, wvT, bv, vTbf,
                                               (int)NTOK, D_MODEL, D_MODEL);

  // 2) fused attention
  dim3 ga(SEQ / 64, NUM_HEADS, BATCH);                  // (32, 16, 2)
  flash_attn_wmma<<<ga, blk, 0, stream>>>(qbuf, kbuf, vTbf, mask, cbuf);

  // 3) output projection (fp32 out)
  gemm_wmma<0, float><<<gg, blk, 0, stream>>>(cbuf, woT, bo, out,
                                              (int)NTOK, D_MODEL, D_MODEL);
}
